// ToyTransformerBlock_32074815767141
// MI455X (gfx1250) — compile-verified, matrix operand order checked by reading, not run on hardware
//
#include <hip/hip_runtime.h>
#include <hip/hip_bf16.h>

// ---------------------------------------------------------------------------
// Types for CDNA5 WMMA (wave32): 16x16x32 bf16 -> f32
// ---------------------------------------------------------------------------
typedef __bf16 v16bf __attribute__((ext_vector_type(16)));
typedef __bf16 v8bf  __attribute__((ext_vector_type(8)));
typedef float  v8f   __attribute__((ext_vector_type(8)));

#define SHUF16(lo, hi) __builtin_shufflevector((lo), (hi), 0,1,2,3,4,5,6,7,8,9,10,11,12,13,14,15)

__device__ __forceinline__ v8f wmma_bf16(v16bf a, v16bf b, v8f c) {
    return __builtin_amdgcn_wmma_f32_16x16x32_bf16(false, a, false, b, (short)0, c, false, false);
}

// ---------------------------------------------------------------------------
// CDNA5 async global->LDS copy (ASYNCcnt-tracked), with safe fallback.
// Builtin prototype (from compiler diagnostic): (AS1 v4i*, AS3 v4i*, imm, imm)
// ---------------------------------------------------------------------------
#if __has_builtin(__builtin_amdgcn_global_load_async_to_lds_b128)
#define USE_ASYNC_LDS 1
#else
#define USE_ASYNC_LDS 0
#endif

typedef int v4i __attribute__((vector_size(16)));
typedef __attribute__((address_space(1))) v4i* as1_v4i;
typedef __attribute__((address_space(3))) v4i* as3_v4i;

__device__ __forceinline__ void cp16_g2l(__bf16* lds_dst, const __bf16* g_src) {
#if USE_ASYNC_LDS
    __builtin_amdgcn_global_load_async_to_lds_b128((as1_v4i)g_src, (as3_v4i)lds_dst, 0, 0);
#else
    *(v8bf*)lds_dst = *(const v8bf*)g_src;
#endif
}

__device__ __forceinline__ void wait_async0() {
#if USE_ASYNC_LDS
#if __has_builtin(__builtin_amdgcn_s_wait_asynccnt)
    __builtin_amdgcn_s_wait_asynccnt(0);
#else
    asm volatile("s_wait_asynccnt 0x0" ::: "memory");
#endif
#endif
}

#define EMB    1024
#define HEADS  16
#define DH     64
#define SEQ    2048
#define BATCH  4
#define MROWS  (BATCH * SEQ)   // 8192

// ---------------------------------------------------------------------------
// Weight transpose + fp32 -> bf16 convert:  Wt[n*K + k] = bf16(W[k*N + n])
// ---------------------------------------------------------------------------
__global__ void wtrans_kernel(const float* __restrict__ W, __bf16* __restrict__ Wt,
                              int K, int N) {
    size_t idx = (size_t)blockIdx.x * blockDim.x + threadIdx.x;
    size_t k = idx / (size_t)N;
    size_t n = idx % (size_t)N;
    if (k < (size_t)K) Wt[n * (size_t)K + k] = (__bf16)W[k * (size_t)N + n];
}

// ---------------------------------------------------------------------------
// LayerNorm over last dim (1024), output bf16. One block (256 thr) per row.
// ---------------------------------------------------------------------------
__global__ __launch_bounds__(256) void ln_kernel(const float* __restrict__ X,
                                                 const float* __restrict__ g,
                                                 const float* __restrict__ bta,
                                                 __bf16* __restrict__ out) {
    int row = blockIdx.x;
    const float* x = X + (size_t)row * EMB;
    int t = threadIdx.x;
    float vals[4];
    float s = 0.f, s2 = 0.f;
#pragma unroll
    for (int i = 0; i < 4; i++) {
        float v = x[t + 256 * i];
        vals[i] = v; s += v; s2 += v * v;
    }
#pragma unroll
    for (int off = 16; off >= 1; off >>= 1) {
        s  += __shfl_xor(s,  off);
        s2 += __shfl_xor(s2, off);
    }
    __shared__ float ss[8], ss2[8];
    int wid = t >> 5, lane = t & 31;
    if (lane == 0) { ss[wid] = s; ss2[wid] = s2; }
    __syncthreads();
    if (t == 0) {
        float a = 0.f, a2 = 0.f;
#pragma unroll
        for (int i = 0; i < 8; i++) { a += ss[i]; a2 += ss2[i]; }
        ss[0] = a; ss2[0] = a2;
    }
    __syncthreads();
    float mean = ss[0] * (1.0f / EMB);
    float var  = ss2[0] * (1.0f / EMB) - mean * mean;
    float rstd = rsqrtf(var + 1e-5f);
    __bf16* o = out + (size_t)row * EMB;
#pragma unroll
    for (int i = 0; i < 4; i++) {
        int c = t + 256 * i;
        o[c] = (__bf16)((vals[i] - mean) * rstd * g[c] + bta[c]);
    }
}

// ---------------------------------------------------------------------------
// Tiled WMMA GEMM:  C[M,N] = A[M,K](bf16,row) x Bt[N,K](bf16, K-contiguous)
// Block tile 128x256x32, 256 threads = 8 waves in 2(M) x 4(N); each wave owns
// a 64x64 patch = 4x4 WMMA accumulators (16 WMMAs : 16 ds_load_b128 per step).
// Double-buffered LDS fed by async global->LDS copies.
// Epilogue modes:
//   0: outb = bf16(acc)
//   1: outf = acc + bias[n] + resid[m*N+n]
//   2: outb = bf16(gelu_exact(acc + bias[n]))
// ---------------------------------------------------------------------------
#define BM 128
#define BN 256
#define BK 32

template <int MODE>
__global__ __launch_bounds__(256) void gemm_bf16_kernel(
    const __bf16* __restrict__ A, const __bf16* __restrict__ Bt,
    int M, int N, int K,
    const float* __restrict__ bias, const float* __restrict__ resid,
    float* __restrict__ outf, __bf16* __restrict__ outb) {

    __shared__ __align__(16) __bf16 As[2][BM * BK];
    __shared__ __align__(16) __bf16 Bs[2][BN * BK];

    int tid  = threadIdx.x;
    int m0   = blockIdx.y * BM;
    int n0   = blockIdx.x * BN;
    int wave = tid >> 5;
    int lane = tid & 31;
    int lrow = lane & 15;
    int hi   = lane >> 4;
    int waveM = wave >> 2;   // 0..1
    int waveN = wave & 3;    // 0..3

    v8f acc[4][4];
    const v8f vzero = {0.f, 0.f, 0.f, 0.f, 0.f, 0.f, 0.f, 0.f};
#pragma unroll
    for (int i = 0; i < 4; i++)
#pragma unroll
        for (int j = 0; j < 4; j++) acc[i][j] = vzero;

    // stage tile kt into LDS buffer `buf` (async if available)
    auto stage = [&](int kt, int buf) {
        int k0 = kt * BK;
#pragma unroll
        for (int i = 0; i < 2; i++) {                 // A: 512 chunks / 256 thr
            int c  = tid + i * 256;
            int r  = c >> 2;                          // 0..127
            int kc = (c & 3) * 8;                     // 0,8,16,24
            cp16_g2l(&As[buf][r * BK + kc], &A[(size_t)(m0 + r) * K + k0 + kc]);
        }
#pragma unroll
        for (int i = 0; i < 4; i++) {                 // B: 1024 chunks / 256 thr
            int c  = tid + i * 256;
            int r  = c >> 2;                          // 0..255
            int kc = (c & 3) * 8;
            cp16_g2l(&Bs[buf][r * BK + kc], &Bt[(size_t)(n0 + r) * K + k0 + kc]);
        }
    };

    int nk = K / BK;
    stage(0, 0);
    wait_async0();
    __syncthreads();

    for (int kt = 0; kt < nk; kt++) {
        int buf = kt & 1;
        if (kt + 1 < nk) stage(kt + 1, buf ^ 1);      // prefetch next tile

        v16bf afrag[4];
#pragma unroll
        for (int i = 0; i < 4; i++) {
            int r = waveM * 64 + i * 16 + lrow;
            v8bf lo = *(const v8bf*)&As[buf][r * BK + hi * 8];
            v8bf h8 = *(const v8bf*)&As[buf][r * BK + 16 + hi * 8];
            afrag[i] = SHUF16(lo, h8);
        }
#pragma unroll
        for (int j = 0; j < 4; j++) {
            int r = waveN * 64 + j * 16 + lrow;
            v8bf lo = *(const v8bf*)&Bs[buf][r * BK + hi * 8];
            v8bf h8 = *(const v8bf*)&Bs[buf][r * BK + 16 + hi * 8];
            v16bf bfrag = SHUF16(lo, h8);
#pragma unroll
            for (int i = 0; i < 4; i++)
                acc[i][j] = wmma_bf16(afrag[i], bfrag, acc[i][j]);
        }

        wait_async0();
        __syncthreads();
    }

    // epilogue
#pragma unroll
    for (int i = 0; i < 4; i++) {
#pragma unroll
        for (int j = 0; j < 4; j++) {
            int n = n0 + waveN * 64 + j * 16 + lrow;
#pragma unroll
            for (int v = 0; v < 8; v++) {
                int m = m0 + waveM * 64 + i * 16 + v + hi * 8;
                size_t off = (size_t)m * N + n;
                float val = acc[i][j][v];
                if constexpr (MODE == 0) {
                    outb[off] = (__bf16)val;
                } else if constexpr (MODE == 1) {
                    outf[off] = val + bias[n] + resid[off];
                } else {
                    float t2 = val + bias[n];
                    outb[off] = (__bf16)(0.5f * t2 * (1.0f + erff(t2 * 0.70710678118654752f)));
                }
            }
        }
    }
}

// ---------------------------------------------------------------------------
// Flash attention, causal. One block = (b, h, 64-query tile), 128 thr = 4 waves.
// Each wave owns 16 queries. KV blocks of 32. All WMMA.
// ---------------------------------------------------------------------------
__global__ __launch_bounds__(128) void attn_kernel(
    const __bf16* __restrict__ Q, const __bf16* __restrict__ Kb,
    const __bf16* __restrict__ Vb, __bf16* __restrict__ O) {

    __shared__ __align__(16) __bf16 Pl[4 * 16 * 32];   // per-wave P tiles
    __shared__ __align__(16) __bf16 Vt[64 * 32];       // V transposed: [d][key]

    int qb = blockIdx.x;       // 0..31 (64-query tiles)
    int h  = blockIdx.y;       // 0..15
    int b  = blockIdx.z;       // 0..3
    int tid = threadIdx.x;
    int w = tid >> 5, lane = tid & 31, lrow = lane & 15, hi = lane >> 4;
    int qbase = qb * 64 + w * 16;
    size_t base = ((size_t)b * SEQ) * EMB + (size_t)h * DH;

    // Q fragments (16 queries x 64 d, K-dim split into 2 x 32)
    v16bf qf[2];
#pragma unroll
    for (int ks = 0; ks < 2; ks++) {
        const __bf16* p = Q + base + (size_t)(qbase + lrow) * EMB + ks * 32;
        v8bf lo = *(const v8bf*)(p + hi * 8);
        v8bf h8 = *(const v8bf*)(p + 16 + hi * 8);
        qf[ks] = SHUF16(lo, h8);
    }

    float mrun[8], lrun[8];
    v8f accv[4];
    const v8f vzero = {0.f, 0.f, 0.f, 0.f, 0.f, 0.f, 0.f, 0.f};
#pragma unroll
    for (int v = 0; v < 8; v++) { mrun[v] = -__builtin_inff(); lrun[v] = 0.f; }
#pragma unroll
    for (int n = 0; n < 4; n++) accv[n] = vzero;

    const float scale = 0.125f;   // 1/sqrt(64)
    int nblk = qb * 2 + 2;        // cover keys up to qb*64+63 (uniform over waves)

    for (int t = 0; t < nblk; t++) {
        int kv0 = t * 32;

        // K fragments: 2 key-tiles x 2 d-steps
        v16bf kf[2][2];
#pragma unroll
        for (int j = 0; j < 2; j++) {
#pragma unroll
            for (int ks = 0; ks < 2; ks++) {
                const __bf16* p = Kb + base + (size_t)(kv0 + j * 16 + lrow) * EMB + ks * 32;
                v8bf lo = *(const v8bf*)(p + hi * 8);
                v8bf h8 = *(const v8bf*)(p + 16 + hi * 8);
                kf[j][ks] = SHUF16(lo, h8);
            }
        }

        // scores = Q x K^T  (16 x 32)
        v8f s[2];
#pragma unroll
        for (int j = 0; j < 2; j++) {
            s[j] = vzero;
            s[j] = wmma_bf16(qf[0], kf[j][0], s[j]);
            s[j] = wmma_bf16(qf[1], kf[j][1], s[j]);
        }

        // causal mask + scale
        float ps[2][8];
#pragma unroll
        for (int j = 0; j < 2; j++) {
            int key = kv0 + j * 16 + lrow;
#pragma unroll
            for (int v = 0; v < 8; v++) {
                int qi = qbase + v + hi * 8;
                float val = s[j][v] * scale;
                ps[j][v] = (key <= qi) ? val : -__builtin_inff();
            }
        }

        // row max across the 16-lane half (rows m = v + 8*hi)
        float r[8];
#pragma unroll
        for (int v = 0; v < 8; v++) r[v] = fmaxf(ps[0][v], ps[1][v]);
#pragma unroll
        for (int off = 8; off >= 1; off >>= 1)
#pragma unroll
            for (int v = 0; v < 8; v++) r[v] = fmaxf(r[v], __shfl_xor(r[v], off));

        float mnew[8], corr[8], psum[8];
#pragma unroll
        for (int v = 0; v < 8; v++) {
            mnew[v] = fmaxf(mrun[v], r[v]);
            corr[v] = __expf(mrun[v] - mnew[v]);   // exp(-inf)=0 on first block
        }
#pragma unroll
        for (int j = 0; j < 2; j++)
#pragma unroll
            for (int v = 0; v < 8; v++) ps[j][v] = __expf(ps[j][v] - mnew[v]);
#pragma unroll
        for (int v = 0; v < 8; v++) psum[v] = ps[0][v] + ps[1][v];
#pragma unroll
        for (int off = 8; off >= 1; off >>= 1)
#pragma unroll
            for (int v = 0; v < 8; v++) psum[v] += __shfl_xor(psum[v], off);
#pragma unroll
        for (int v = 0; v < 8; v++) {
            lrun[v] = lrun[v] * corr[v] + psum[v];
            mrun[v] = mnew[v];
        }
#pragma unroll
        for (int n = 0; n < 4; n++)
#pragma unroll
            for (int v = 0; v < 8; v++) accv[n][v] *= corr[v];

        __syncthreads();   // protect Pl/Vt from previous iteration's readers

        // stage P (C-layout -> A-layout via LDS), bf16
#pragma unroll
        for (int j = 0; j < 2; j++)
#pragma unroll
            for (int v = 0; v < 8; v++)
                Pl[w * 512 + (v + hi * 8) * 32 + j * 16 + lrow] = (__bf16)ps[j][v];

        // stage V transposed: Vt[d][key], cooperatively across 128 threads
#pragma unroll
        for (int i = 0; i < 2; i++) {
            int c   = tid + i * 128;       // 0..255 chunks of 8 d-values
            int key = c >> 3;              // 0..31
            int dc  = (c & 7) * 8;         // 0..56
            v8bf ch = *(const v8bf*)(Vb + base + (size_t)(kv0 + key) * EMB + dc);
#pragma unroll
            for (int e = 0; e < 8; e++) Vt[(dc + e) * 32 + key] = ch[e];
        }
        __syncthreads();

        // P fragment (16 x 32 over keys)
        v16bf pf;
        {
            const __bf16* p = &Pl[w * 512 + lrow * 32];
            v8bf lo = *(const v8bf*)(p + hi * 8);
            v8bf h8 = *(const v8bf*)(p + 16 + hi * 8);
            pf = SHUF16(lo, h8);
        }
        // ctx += P x V  (4 d-tiles of 16)
#pragma unroll
        for (int n = 0; n < 4; n++) {
            const __bf16* p = &Vt[(n * 16 + lrow) * 32];
            v8bf lo = *(const v8bf*)(p + hi * 8);
            v8bf h8 = *(const v8bf*)(p + 16 + hi * 8);
            v16bf vf = SHUF16(lo, h8);
            accv[n] = wmma_bf16(pf, vf, accv[n]);
        }
    }

    // normalize and store ctx (bf16) into [B,S,H*DH]
#pragma unroll
    for (int n = 0; n < 4; n++)
#pragma unroll
        for (int v = 0; v < 8; v++) {
            float val = accv[n][v] / lrun[v];
            O[base + (size_t)(qbase + v + hi * 8) * EMB + n * 16 + lrow] = (__bf16)val;
        }
}

// ---------------------------------------------------------------------------
// Host-side orchestration
// ---------------------------------------------------------------------------
extern "C" void kernel_launch(void* const* d_in, const int* in_sizes, int n_in,
                              void* d_out, int out_size, void* d_ws, size_t ws_size,
                              hipStream_t stream) {
    const float* x     = (const float*)d_in[0];
    const float* Wq    = (const float*)d_in[1];
    const float* Wk    = (const float*)d_in[2];
    const float* Wv    = (const float*)d_in[3];
    const float* Wo    = (const float*)d_in[4];
    const float* bo    = (const float*)d_in[5];
    const float* W1    = (const float*)d_in[6];
    const float* b1    = (const float*)d_in[7];
    const float* W2    = (const float*)d_in[8];
    const float* b2    = (const float*)d_in[9];
    const float* g1    = (const float*)d_in[10];
    const float* beta1 = (const float*)d_in[11];
    const float* g2    = (const float*)d_in[12];
    const float* beta2 = (const float*)d_in[13];
    float* out = (float*)d_out;

    char* ws = (char*)d_ws;
    size_t off = 0;
    auto carve = [&](size_t bytes) -> void* {
        void* p = ws + off;
        off += (bytes + 255) & ~(size_t)255;
        return p;
    };

    const size_t M = MROWS;
    __bf16* ln1  = (__bf16*)carve(M * EMB * 2);
    __bf16* qbuf = (__bf16*)carve(M * EMB * 2);
    __bf16* kbuf = (__bf16*)carve(M * EMB * 2);
    __bf16* vbuf = (__bf16*)carve(M * EMB * 2);
    __bf16* ctxb = (__bf16*)carve(M * EMB * 2);
    float*  hbuf = (float*) carve(M * EMB * 4);
    __bf16* WqT  = (__bf16*)carve((size_t)EMB * EMB * 2);
    __bf16* WkT  = (__bf16*)carve((size_t)EMB * EMB * 2);
    __bf16* WvT  = (__bf16*)carve((size_t)EMB * EMB * 2);
    __bf16* WoT  = (__bf16*)carve((size_t)EMB * EMB * 2);
    __bf16* W1T  = (__bf16*)carve((size_t)EMB * 4 * EMB * 2);
    __bf16* W2T  = (__bf16*)carve((size_t)4 * EMB * EMB * 2);
    // aliased reuse (dead by the time they're rewritten):
    __bf16* ln2  = ln1;              // ln1 dead after QKV gemms
    __bf16* a1   = qbuf;             // q/k/v/ctx (64MB contig) dead after Wo gemm

    // 1) weight transpose+convert to bf16 column-major
    {
        int n1 = EMB * EMB;
        wtrans_kernel<<<n1 / 256, 256, 0, stream>>>(Wq, WqT, EMB, EMB);
        wtrans_kernel<<<n1 / 256, 256, 0, stream>>>(Wk, WkT, EMB, EMB);
        wtrans_kernel<<<n1 / 256, 256, 0, stream>>>(Wv, WvT, EMB, EMB);
        wtrans_kernel<<<n1 / 256, 256, 0, stream>>>(Wo, WoT, EMB, EMB);
        int n2 = EMB * 4 * EMB;
        wtrans_kernel<<<n2 / 256, 256, 0, stream>>>(W1, W1T, EMB, 4 * EMB);
        wtrans_kernel<<<n2 / 256, 256, 0, stream>>>(W2, W2T, 4 * EMB, EMB);
    }

    // 2) ln1 = LN(x; g1, beta1) -> bf16
    ln_kernel<<<(int)M, 256, 0, stream>>>(x, g1, beta1, ln1);

    // 3) Q/K/V projections
    dim3 gP(EMB / BN, (int)M / BM);
    gemm_bf16_kernel<0><<<gP, 256, 0, stream>>>(ln1, WqT, (int)M, EMB, EMB, nullptr, nullptr, nullptr, qbuf);
    gemm_bf16_kernel<0><<<gP, 256, 0, stream>>>(ln1, WkT, (int)M, EMB, EMB, nullptr, nullptr, nullptr, kbuf);
    gemm_bf16_kernel<0><<<gP, 256, 0, stream>>>(ln1, WvT, (int)M, EMB, EMB, nullptr, nullptr, nullptr, vbuf);

    // 4) causal flash attention
    attn_kernel<<<dim3(SEQ / 64, HEADS, BATCH), 128, 0, stream>>>(qbuf, kbuf, vbuf, ctxb);

    // 5) h = x + ctx @ Wo + bo   (f32)
    gemm_bf16_kernel<1><<<gP, 256, 0, stream>>>(ctxb, WoT, (int)M, EMB, EMB, bo, x, hbuf, nullptr);

    // 6) ln2 = LN(h; g2, beta2) -> bf16
    ln_kernel<<<(int)M, 256, 0, stream>>>(hbuf, g2, beta2, ln2);

    // 7) a1 = gelu(ln2 @ W1 + b1) -> bf16
    dim3 gF(4 * EMB / BN, (int)M / BM);
    gemm_bf16_kernel<2><<<gF, 256, 0, stream>>>(ln2, W1T, (int)M, 4 * EMB, EMB, b1, nullptr, nullptr, a1);

    // 8) out = h + a1 @ W2 + b2  (f32)
    gemm_bf16_kernel<1><<<gP, 256, 0, stream>>>(a1, W2T, (int)M, EMB, 4 * EMB, b2, hbuf, out, nullptr);

    (void)in_sizes; (void)n_in; (void)out_size; (void)ws_size;
}
